// SigModuleParallel_1614907703713
// MI455X (gfx1250) — matrix operand order repaired and checked by reading, not compiled, not run on hardware
//
#include <hip/hip_runtime.h>
#include <hip/hip_bf16.h>

typedef __attribute__((ext_vector_type(16))) __bf16 v16bf;
typedef __attribute__((ext_vector_type(8)))  __bf16 v8bf;
typedef __attribute__((ext_vector_type(8)))  float  v8f;

#define NB    240000      // total windows = 32*25*300
#define Tn    300
#define Jn    25
#define Cn    16
#define Dn    17          // signature path dim (time + 16 channels)
#define KSIG  306         // 17 + 17*17
#define KPAD  320         // padded K for 10 k-steps of 32
#define KROW  328         // LDS row stride (bank-conflict padding, 16B aligned)
#define MT    32          // windows per workgroup tile
#define TILES 7500        // NB / MT
#define ODIM  64

__device__ __forceinline__ __bf16 f2bf(float f) {
    unsigned u = __builtin_bit_cast(unsigned, f);
    unsigned r = u + 0x7FFFu + ((u >> 16) & 1u);   // round-to-nearest-even
    return __builtin_bit_cast(__bf16, (unsigned short)(r >> 16));
}

__global__ __launch_bounds__(256) void sig_wmma_kernel(
    const float* __restrict__ x,     // [32,16,25,300]
    const float* __restrict__ W,     // [64,306]
    const float* __restrict__ bvec,  // [64]
    float* __restrict__ out)         // [32,64,25,300]
{
    __shared__ float qArr[5 * Dn * MT];
    __shared__ float dArr[5 * Dn * MT];
    __shared__ float pend[Dn * MT];
    __shared__ __attribute__((aligned(16))) __bf16 sigL[MT * KROW];
    __shared__ float outT[ODIM * MT];

    const int tid  = threadIdx.x;
    const int lane = tid & 31;
    const int wv   = tid >> 5;          // 8 waves
    const int ms   = wv >> 2;           // M-subtile 0..1
    const int nt   = wv & 3;            // N-tile 0..3

    // ---- Load this wave's B-fragments of W (bf16) into registers, once ----
    // B k order chosen here (K x N, contiguous, same as A): lane l -> col n = l&15 ; K = kk*32 + (l<16?0:16) + j
    v16bf bf[10];
    {
        const int n     = nt * 16 + (lane & 15);
        const int kbase = (lane & 16) ? 16 : 0;
        #pragma unroll
        for (int kk = 0; kk < 10; ++kk) {
            v16bf t;
            #pragma unroll
            for (int j = 0; j < 16; ++j) {
                int k = kk * 32 + kbase + j;
                t[j] = f2bf(k < KSIG ? W[n * KSIG + k] : 0.f);
            }
            bf[kk] = t;
        }
    }
    const float bias = bvec[nt * 16 + (lane & 15)];

    // ---- Zero the sig K-pad region (k = 306..319) once; never rewritten ----
    for (int e = tid; e < MT * (KPAD - KSIG); e += 256) {
        int w = e / (KPAD - KSIG);
        int k = KSIG + e % (KPAD - KSIG);
        sigL[w * KROW + k] = f2bf(0.f);
    }

    for (int tile = blockIdx.x; tile < TILES; tile += gridDim.x) {
        const int base = tile * MT;
        __syncthreads();

        // ---- Phase 1: increments d_k, q_k = P_{k-1} + 0.5*d_k, endpoint ----
        for (int task = tid; task < Dn * MT; task += 256) {
            const int w   = task & (MT - 1);
            const int dim = task >> 5;
            float p[5];
            if (dim == 0) {             // time channel: linspace(0,1,5)
                p[0] = 0.f; p[1] = 0.25f; p[2] = 0.5f; p[3] = 0.75f; p[4] = 1.f;
            } else {
                const int b   = base + w;
                const int nI  = b / (Jn * Tn);
                const int rem = b - nI * (Jn * Tn);
                const int jI  = rem / Tn;
                const int tI  = rem - jI * Tn;
                const float* xp = x + ((nI * Cn + (dim - 1)) * Jn + jI) * Tn;
                #pragma unroll
                for (int k = 0; k < 5; ++k) {
                    int tt = tI + k - 2;
                    tt = tt < 0 ? 0 : (tt > Tn - 1 ? Tn - 1 : tt);
                    p[k] = xp[tt];
                }
            }
            float prev = 0.f;
            #pragma unroll
            for (int k = 0; k < 5; ++k) {
                float dk = p[k] - prev;
                dArr[(k * Dn + dim) * MT + w] = dk;
                qArr[(k * Dn + dim) * MT + w] = prev + 0.5f * dk;
                prev = p[k];
            }
            pend[dim * MT + w] = p[4];
        }
        __syncthreads();

        // ---- Phase 2: build bf16 signature tile rows in LDS ----
        // sig[w][i] = P4[i] (level 1); sig[w][17 + i*17 + j] = sum_k q_k[i]*d_k[j]
        for (int task = tid; task < Dn * MT; task += 256) {
            const int w = task & (MT - 1);
            const int i = task >> 5;
            __bf16* row = &sigL[w * KROW];
            row[i] = f2bf(pend[i * MT + w]);
            float qi[5];
            #pragma unroll
            for (int k = 0; k < 5; ++k) qi[k] = qArr[(k * Dn + i) * MT + w];
            for (int j = 0; j < Dn; ++j) {
                float s = 0.f;
                #pragma unroll
                for (int k = 0; k < 5; ++k) s += qi[k] * dArr[(k * Dn + j) * MT + w];
                row[Dn + i * Dn + j] = f2bf(s);
            }
        }
        __syncthreads();

        // ---- Phase 3: WMMA  (sig 16xK tile) x (W^T Kx16) -> 16x16 f32 ----
        {
            const int mrow = ms * 16 + (lane & 15);
            const __bf16* arow = &sigL[mrow * KROW + ((lane & 16) ? 16 : 0)];
            v8f acc = {};
            #pragma unroll
            for (int kk = 0; kk < 10; ++kk) {
                v8bf lo = *(const v8bf*)(arow + kk * 32);        // K = kk*32+off+0..7
                v8bf hi = *(const v8bf*)(arow + kk * 32 + 8);    // K = kk*32+off+8..15
                v16bf a;
                #pragma unroll
                for (int j = 0; j < 8; ++j) { a[j] = lo[j]; a[j + 8] = hi[j]; }
                acc = __builtin_amdgcn_wmma_f32_16x16x32_bf16(
                          false, a, false, bf[kk], (short)0, acc, false, false);
            }
            // C/D layout: VGPR r, lane l -> row = r + (l<16?0:8), col = l&15
            const int o     = nt * 16 + (lane & 15);
            const int wbase = ms * 16 + ((lane & 16) ? 8 : 0);
            #pragma unroll
            for (int r = 0; r < 8; ++r)
                outT[o * MT + wbase + r] = acc[r] + bias;
        }
        __syncthreads();

        // ---- Phase 4: coalesced transpose-store to out[n][o][j][t] ----
        for (int e = tid; e < ODIM * MT; e += 256) {
            const int o   = e >> 5;
            const int w   = e & 31;
            const int b   = base + w;
            const int nI  = b / (Jn * Tn);
            const int rem = b - nI * (Jn * Tn);
            const int jI  = rem / Tn;
            const int tI  = rem - jI * Tn;
            out[((nI * ODIM + o) * Jn + jI) * Tn + tI] = outT[e];
        }
    }
}

extern "C" void kernel_launch(void* const* d_in, const int* in_sizes, int n_in,
                              void* d_out, int out_size, void* d_ws, size_t ws_size,
                              hipStream_t stream) {
    const float* x    = (const float*)d_in[0];   // [32,16,25,300]
    const float* W    = (const float*)d_in[1];   // [64,306]
    const float* bvec = (const float*)d_in[2];   // [64]
    float* out = (float*)d_out;                  // [32,64,25,300]
    (void)in_sizes; (void)n_in; (void)out_size; (void)d_ws; (void)ws_size;
    sig_wmma_kernel<<<1250, 256, 0, stream>>>(x, W, bvec, out);
}
